// SmolVLAPrefixModel_23613730194152
// MI455X (gfx1250) — matrix operand order repaired and checked by reading, not compiled, not run on hardware
//
#include <hip/hip_runtime.h>
#include <hip/hip_bf16.h>
#include <math.h>

// ---------------- model constants ----------------
#define L_     16
#define HID_   960
#define NH_    15
#define NKV_   5
#define HD_    64
#define INTER_ 2560
#define NV_    192
#define NT_    48
#define PRE_   241
#define SD_    32
#define B_     16
#define GROUPS_ (NH_ / NKV_)          // 3
#define MROWS_ (B_ * PRE_)            // 3856
#define BIGNEG (-3.4028235e38f)

// ---------------- WMMA types ----------------
typedef __attribute__((ext_vector_type(16))) __bf16    v16bf;
typedef __attribute__((ext_vector_type(16))) _Float16  v16h;
typedef __attribute__((ext_vector_type(8)))  float     v8f;
typedef __attribute__((ext_vector_type(4)))  unsigned int uint4v;

union FragBF { uint4v q[2]; unsigned short us[16]; v16bf v; };
union FragH  { uint4v q[2]; unsigned short us[16]; _Float16 h[16]; v16h v; };
union H8     { uint4v q; _Float16 h[8]; };

__device__ __forceinline__ v8f v8f_zero() {
  v8f z;
  #pragma unroll
  for (int i = 0; i < 8; ++i) z[i] = 0.0f;
  return z;
}

__device__ __forceinline__ uint4v u4_zero() {
  uint4v z;
  #pragma unroll
  for (int i = 0; i < 4; ++i) z[i] = 0u;
  return z;
}

__device__ __forceinline__ unsigned short f32_to_bf16(float f) {
  unsigned int u = __float_as_uint(f);
  unsigned int r = u + 0x7FFFu + ((u >> 16) & 1u);   // round to nearest even
  return (unsigned short)(r >> 16);
}

__device__ __forceinline__ int cumv(int i) {          // cumsum of att_marks
  return (i >= NV_ + NT_) ? (i - (NV_ + NT_) + 1) : 0;
}

// ================= embedding / x assembly =================
__global__ __launch_bounds__(256)
void build_x_kernel(const float* __restrict__ vision, const int* __restrict__ tok,
                    const float* __restrict__ state, const float* __restrict__ embed,
                    const float* __restrict__ state_w, const float* __restrict__ state_b,
                    float* __restrict__ x) {
  int row = blockIdx.x;             // 0..MROWS_-1
  int b = row / PRE_, i = row % PRE_;
  const float sq = 30.98386676965934f;  // sqrt(960)
  for (int c = threadIdx.x; c < HID_; c += 256) {
    float val;
    if (i < NV_) {
      val = vision[((long)b * NV_ + i) * HID_ + c];
    } else if (i < NV_ + NT_) {
      int t = tok[b * NT_ + (i - NV_)];
      val = embed[(long)t * HID_ + c] * sq;
    } else {
      float a = state_b[c];
      #pragma unroll
      for (int s = 0; s < SD_; ++s) a += state[b * SD_ + s] * state_w[s * HID_ + c];
      val = a;
    }
    x[(long)row * HID_ + c] = val;
  }
}

// ================= pad & positions =================
__global__ void padpos_kernel(const unsigned char* __restrict__ lang_mask,
                              int* __restrict__ pad, int* __restrict__ pos) {
  int b = threadIdx.x;
  if (b >= B_) return;
  int c = 0;
  for (int i = 0; i < PRE_; ++i) {
    int pd;
    if (i < NV_) pd = 1;
    else if (i < NV_ + NT_) pd = lang_mask[b * NT_ + (i - NV_)] ? 1 : 0;
    else pd = 1;
    c += pd;
    int p = c - 1; if (p < 0) p = 0;
    pad[b * PRE_ + i] = pd;
    pos[b * PRE_ + i] = p;
  }
}

// ================= RMSNorm (fp32 in -> bf16 out) =================
__global__ __launch_bounds__(256)
void rms_kernel(const float* __restrict__ x, const float* __restrict__ w,
                unsigned short* __restrict__ h16) {
  __shared__ float red[256];
  long row = blockIdx.x;
  float s = 0.f;
  for (int c = threadIdx.x; c < HID_; c += 256) {
    float v = x[row * HID_ + c]; s += v * v;
  }
  red[threadIdx.x] = s; __syncthreads();
  for (int off = 128; off > 0; off >>= 1) {
    if (threadIdx.x < off) red[threadIdx.x] += red[threadIdx.x + off];
    __syncthreads();
  }
  float scale = rsqrtf(red[0] / (float)HID_ + 1e-5f);
  for (int c = threadIdx.x; c < HID_; c += 256)
    h16[row * HID_ + c] = f32_to_bf16(x[row * HID_ + c] * scale * w[c]);
}

// ================= tiled transpose + convert: W[K][N] fp32 -> Wt[N][K] bf16 ===
__global__ __launch_bounds__(256)
void transpose_bf16_kernel(const float* __restrict__ W, unsigned short* __restrict__ Wt,
                           int K, int N) {
  __shared__ float t[32][33];
  int nt = blockIdx.x * 32, kt = blockIdx.y * 32;
  int tx = threadIdx.x & 31, ty = threadIdx.x >> 5;   // 32 x 8
  #pragma unroll
  for (int i = 0; i < 4; ++i) {
    int r = ty + i * 8;
    t[r][tx] = W[(long)(kt + r) * N + nt + tx];
  }
  __syncthreads();
  #pragma unroll
  for (int i = 0; i < 4; ++i) {
    int r = ty + i * 8;
    Wt[(long)(nt + r) * K + kt + tx] = f32_to_bf16(t[tx][r]);
  }
}

// ====== bf16 WMMA GEMM: C = A16[MxK] * Wt16[NxK]^T (+R) ======
// block = 256 threads = 8 waves; block tile 256x64; wave tile 32x64 (2x4 frags)
__global__ __launch_bounds__(256)
void gemm_bf16_wmma(const unsigned short* __restrict__ A16,
                    const unsigned short* __restrict__ Wt16,
                    const float* __restrict__ R, float* __restrict__ C,
                    int M, int N, int K) {
  __shared__ __align__(16) unsigned short As[256 * 40];
  __shared__ __align__(16) unsigned short Ws[64 * 40];

  int tid = threadIdx.x, lane = tid & 31, wid = tid >> 5;
  int wR = wid * 32;                   // wave row base within block tile
  int half = lane >> 4, l16 = lane & 15;
  int m0 = blockIdx.y * 256, n0 = blockIdx.x * 64;

  // staging geometry: 8-element (16B) chunks; A rows clamped (stores are guarded)
  int rBase = tid >> 2;                // 0..63
  int cEl = (tid & 3) * 8;             // 0,8,16,24
  const unsigned short* Aof[4];
  #pragma unroll
  for (int it = 0; it < 4; ++it) {
    int row = m0 + rBase + 64 * it;
    if (row > M - 1) row = M - 1;      // clamp: no predication, garbage rows unused
    Aof[it] = A16 + (long)row * K + cEl;
  }
  const unsigned short* Wof = Wt16 + (long)(n0 + rBase) * K + cEl;

  v8f acc[2][4];
  #pragma unroll
  for (int mi = 0; mi < 2; ++mi)
    #pragma unroll
    for (int ni = 0; ni < 4; ++ni) acc[mi][ni] = v8f_zero();

  // prefetch tile 0
  uint4v pa[4], pw;
  #pragma unroll
  for (int it = 0; it < 4; ++it) pa[it] = *(const uint4v*)(Aof[it]);
  pw = *(const uint4v*)(Wof);

  for (int kk = 0; kk < K; kk += 32) {
    // commit prefetched tile to LDS
    #pragma unroll
    for (int it = 0; it < 4; ++it)
      *(uint4v*)&As[(rBase + 64 * it) * 40 + cEl] = pa[it];
    *(uint4v*)&Ws[rBase * 40 + cEl] = pw;
    __syncthreads();

    // issue next tile's global loads before the math
    int kn = kk + 32;
    if (kn < K) {
      #pragma unroll
      for (int it = 0; it < 4; ++it) pa[it] = *(const uint4v*)(Aof[it] + kn);
      pw = *(const uint4v*)(Wof + kn);
    }

    // hoist the 4 B fragments (shared across both m sub-tiles)
    FragBF bfr[4];
    #pragma unroll
    for (int ni = 0; ni < 4; ++ni) {
      int n = ni * 16 + l16;
      bfr[ni].q[0] = *(const uint4v*)&Ws[n * 40 + half * 8];
      bfr[ni].q[1] = *(const uint4v*)&Ws[n * 40 + 16 + half * 8];
    }
    #pragma unroll
    for (int mi = 0; mi < 2; ++mi) {
      FragBF a;
      int m = wR + mi * 16 + l16;
      a.q[0] = *(const uint4v*)&As[m * 40 + half * 8];
      a.q[1] = *(const uint4v*)&As[m * 40 + 16 + half * 8];
      #pragma unroll
      for (int ni = 0; ni < 4; ++ni) {
        acc[mi][ni] = __builtin_amdgcn_wmma_f32_16x16x32_bf16(
            false, a.v, false, bfr[ni].v, (short)0, acc[mi][ni], false, false);
      }
    }
    __syncthreads();
  }

  #pragma unroll
  for (int mi = 0; mi < 2; ++mi)
    #pragma unroll
    for (int ni = 0; ni < 4; ++ni) {
      int n = n0 + ni * 16 + l16;
      #pragma unroll
      for (int r = 0; r < 8; ++r) {
        int m = m0 + wR + mi * 16 + half * 8 + r;
        if (m < M) {
          float v = acc[mi][ni][r];
          if (R) v += R[(long)m * N + n];
          C[(long)m * N + n] = v;
        }
      }
    }
}

// ==== RoPE: fp32 q/k/v -> f16 attention buffers + fp32 KV cache in d_out ====
__global__ __launch_bounds__(64)
void rope_store_kernel(const float* __restrict__ q, const float* __restrict__ k,
                       const float* __restrict__ v, const int* __restrict__ pos,
                       _Float16* __restrict__ qh, _Float16* __restrict__ kh,
                       _Float16* __restrict__ vh,
                       float* __restrict__ keys_out, float* __restrict__ vals_out) {
  long row = blockIdx.x;                 // b*PRE + s
  int d = threadIdx.x;                   // 0..63
  int d2 = d & 31;
  float invf = __expf(-(float)(2 * d2) * (11.512925464970229f / 64.0f)); // ln(1e5)/64
  float ang = (float)pos[row] * invf;
  float cs = __cosf(ang), sn = __sinf(ang);

  #pragma unroll
  for (int hh = 0; hh < NH_; ++hh) {
    long base = row * (NH_ * HD_) + hh * HD_;
    float xv = q[base + d];
    float xp = q[base + ((d < 32) ? d + 32 : d - 32)];
    float rot = (d < 32) ? -xp : xp;
    qh[base + d] = (_Float16)(xv * cs + rot * sn);
  }
  #pragma unroll
  for (int kh_i = 0; kh_i < NKV_; ++kh_i) {
    long base = row * (NKV_ * HD_) + kh_i * HD_;
    float xv = k[base + d];
    float xp = k[base + ((d < 32) ? d + 32 : d - 32)];
    float rot = (d < 32) ? -xp : xp;
    float rk = xv * cs + rot * sn;
    kh[base + d] = (_Float16)rk;
    keys_out[base + d] = rk;
    float vv = v[base + d];
    vh[base + d] = (_Float16)vv;
    vals_out[base + d] = vv;
  }
}

// ================= attention (f16 WMMA, exact softmax, bf16 out) =================
#define ATTN_SMEM (256*64*2 + 64*256*2 + 16*64*2 + 16*256*4 + 16*256*2 + 128*4 + 16*4 + 16*4)

__global__ __launch_bounds__(128)
void attn_kernel(const _Float16* __restrict__ qh, const _Float16* __restrict__ kh,
                 const _Float16* __restrict__ vh, const int* __restrict__ pad,
                 unsigned short* __restrict__ out16) {
  extern __shared__ char smem[];
  char* sp = smem;
  _Float16* kL = (_Float16*)sp;  sp += 256 * 64 * 2;
  _Float16* vT = (_Float16*)sp;  sp += 64 * 256 * 2;
  _Float16* qL = (_Float16*)sp;  sp += 16 * 64 * 2;
  float*    sL = (float*)sp;     sp += 16 * 256 * 4;
  _Float16* pL = (_Float16*)sp;  sp += 16 * 256 * 2;
  float*    red = (float*)sp;    sp += 128 * 4;
  float*    rowmax = (float*)sp; sp += 16 * 4;
  float*    rowsum = (float*)sp;

  int qb = blockIdx.x, hh = blockIdx.y, b = blockIdx.z;
  int kvh = hh / GROUPS_;
  int tid = threadIdx.x, lane = tid & 31, w = tid >> 5;
  int half = lane >> 4, l16 = lane & 15;

  // stage K ([j][d], b128 copies) and V (transposed [d][j]), zero-padded
  for (int ch = tid; ch < 2048; ch += 128) {
    int j = ch >> 3, c8 = (ch & 7) * 8;
    uint4v kq = u4_zero(), vq = u4_zero();
    if (j < PRE_) {
      long base = ((long)(b * PRE_ + j)) * (NKV_ * HD_) + kvh * HD_ + c8;
      kq = *(const uint4v*)(kh + base);
      vq = *(const uint4v*)(vh + base);
    }
    *(uint4v*)&kL[j * 64 + c8] = kq;
    H8 tv; tv.q = vq;
    #pragma unroll
    for (int e = 0; e < 8; ++e) vT[(c8 + e) * 256 + j] = tv.h[e];
  }
  {
    int m = tid >> 3, c8 = (tid & 7) * 8;      // 128 threads -> all 16x64
    int qi = qb * 16 + m;
    uint4v qq = u4_zero();
    if (qi < PRE_)
      qq = *(const uint4v*)(qh + ((long)(b * PRE_ + qi)) * (NH_ * HD_) + hh * HD_ + c8);
    *(uint4v*)&qL[m * 64 + c8] = qq;
  }
  __syncthreads();

  // scores: each wave handles 4 j-tiles of 16
  for (int jt = w; jt < 16; jt += 4) {
    v8f acc = v8f_zero();
    #pragma unroll
    for (int kk = 0; kk < 64; kk += 32) {
      FragH a, bf;
      a.q[0] = *(const uint4v*)&qL[l16 * 64 + kk + half * 8];
      a.q[1] = *(const uint4v*)&qL[l16 * 64 + kk + 16 + half * 8];
      int jrow = jt * 16 + l16;
      bf.q[0] = *(const uint4v*)&kL[jrow * 64 + kk + half * 8];
      bf.q[1] = *(const uint4v*)&kL[jrow * 64 + kk + 16 + half * 8];
      acc = __builtin_amdgcn_wmma_f32_16x16x32_f16(
          false, a.v, false, bf.v, (short)0, acc, false, false);
    }
    int j = jt * 16 + l16;
    #pragma unroll
    for (int r = 0; r < 8; ++r) {
      int m = half * 8 + r;
      int qi = qb * 16 + m;
      float s = acc[r] * 0.125f;   // HD^-0.5
      bool ok = (j < PRE_) && (qi < PRE_) &&
                (pad[b * PRE_ + j] != 0) && (pad[b * PRE_ + qi] != 0) &&
                (cumv(j) >= cumv(qi));
      sL[m * 256 + j] = ok ? s : BIGNEG;
    }
  }
  __syncthreads();

  // exact softmax over 256 cols (padding cols hold BIGNEG -> 0)
  int row = tid >> 3, part = tid & 7;
  float mx = BIGNEG;
  for (int c = part * 32; c < part * 32 + 32; ++c) mx = fmaxf(mx, sL[row * 256 + c]);
  red[row * 8 + part] = mx; __syncthreads();
  if (part == 0) {
    float m2 = red[row * 8];
    for (int i = 1; i < 8; ++i) m2 = fmaxf(m2, red[row * 8 + i]);
    rowmax[row] = m2;
  }
  __syncthreads();
  float rm = rowmax[row];
  float sm = 0.f;
  for (int c = part * 32; c < part * 32 + 32; ++c) sm += __expf(sL[row * 256 + c] - rm);
  red[row * 8 + part] = sm; __syncthreads();
  if (part == 0) {
    float t = 0.f;
    for (int i = 0; i < 8; ++i) t += red[row * 8 + i];
    rowsum[row] = t;
  }
  __syncthreads();
  float inv = 1.0f / rowsum[row];
  for (int c = part * 32; c < part * 32 + 32; ++c)
    pL[row * 256 + c] = (_Float16)(__expf(sL[row * 256 + c] - rm) * inv);
  __syncthreads();

  // out = P[16x256] @ V[256x64]; wave w owns 16 output columns
  int nb = w * 16;
  v8f acc = v8f_zero();
  #pragma unroll
  for (int kt = 0; kt < 8; ++kt) {
    int kbase = kt * 32;
    FragH a, bf;
    a.q[0] = *(const uint4v*)&pL[l16 * 256 + kbase + half * 8];
    a.q[1] = *(const uint4v*)&pL[l16 * 256 + kbase + 16 + half * 8];
    int n = nb + l16;
    bf.q[0] = *(const uint4v*)&vT[n * 256 + kbase + half * 8];
    bf.q[1] = *(const uint4v*)&vT[n * 256 + kbase + 16 + half * 8];
    acc = __builtin_amdgcn_wmma_f32_16x16x32_f16(
        false, a.v, false, bf.v, (short)0, acc, false, false);
  }
  #pragma unroll
  for (int r = 0; r < 8; ++r) {
    int m = half * 8 + r;
    int qi = qb * 16 + m;
    if (qi < PRE_)
      out16[((long)(b * PRE_ + qi)) * (NH_ * HD_) + hh * HD_ + nb + l16] = f32_to_bf16(acc[r]);
  }
}

// ================= SiLU(g)*u -> bf16 =================
__global__ __launch_bounds__(256)
void silu_mul_kernel(const float* __restrict__ g, const float* __restrict__ u,
                     unsigned short* __restrict__ t16, long n) {
  long i = (long)blockIdx.x * 256 + threadIdx.x;
  if (i < n) {
    float x = g[i];
    float t = x / (1.0f + __expf(-x));
    t16[i] = f32_to_bf16(t * u[i]);
  }
}

// ================= host orchestration =================
extern "C" void kernel_launch(void* const* d_in, const int* in_sizes, int n_in,
                              void* d_out, int out_size, void* d_ws, size_t ws_size,
                              hipStream_t stream) {
  (void)in_sizes; (void)n_in; (void)out_size; (void)ws_size;

  const float*        vision  = (const float*)d_in[0];
  const int*          tok     = (const int*)d_in[1];
  const unsigned char* mask   = (const unsigned char*)d_in[2];
  const float*        state   = (const float*)d_in[3];
  const float*        embed   = (const float*)d_in[4];
  const float*        state_w = (const float*)d_in[5];
  const float*        state_b = (const float*)d_in[6];
  const float*        Wq      = (const float*)d_in[7];
  const float*        Wk      = (const float*)d_in[8];
  const float*        Wv      = (const float*)d_in[9];
  const float*        Wo      = (const float*)d_in[10];
  const float*        Wg      = (const float*)d_in[11];
  const float*        Wu      = (const float*)d_in[12];
  const float*        Wd      = (const float*)d_in[13];
  const float*        ln1     = (const float*)d_in[14];
  const float*        ln2     = (const float*)d_in[15];

  // workspace layout (float units; 16-bit buffers take half)
  float* ws = (float*)d_ws;
  const long SZ_X  = (long)MROWS_ * HID_;          // 3,701,760
  const long SZ_KV = (long)MROWS_ * NKV_ * HD_;    // 1,233,920
  const long SZ_I  = (long)MROWS_ * INTER_;        // 9,871,360
  const long SZ_WT = (long)INTER_ * HID_;          // 2,457,600 (largest weight)

  float* x   = ws;  ws += SZ_X;
  float* qbf = ws;  ws += SZ_X;
  float* kbf = ws;  ws += SZ_KV;
  float* vbf = ws;  ws += SZ_KV;
  float* gbf = ws;  ws += SZ_I;
  float* ubf = ws;  ws += SZ_I;
  unsigned short* hb16 = (unsigned short*)ws;  ws += SZ_X / 2;
  unsigned short* ab16 = (unsigned short*)ws;  ws += SZ_X / 2;
  unsigned short* tb16 = (unsigned short*)ws;  ws += SZ_I / 2;
  unsigned short* wtb  = (unsigned short*)ws;  ws += SZ_WT / 2;
  _Float16* qh = (_Float16*)ws;  ws += SZ_X / 2;
  _Float16* kh = (_Float16*)ws;  ws += SZ_KV / 2;
  _Float16* vh = (_Float16*)ws;  ws += SZ_KV / 2;
  int* pad = (int*)ws;  ws += MROWS_;
  int* pos = (int*)ws;

  float* keys_out = (float*)d_out;
  const long BPN = (long)B_ * PRE_ * NKV_ * HD_;   // per-layer cache size
  float* vals_out = keys_out + (long)L_ * BPN;

  build_x_kernel<<<MROWS_, 256, 0, stream>>>(vision, tok, state, embed, state_w, state_b, x);
  padpos_kernel<<<1, 32, 0, stream>>>(mask, pad, pos);

  const dim3 gB(256);
  const int gy = (MROWS_ + 255) / 256;             // 16

  for (int layer = 0; layer < L_; ++layer) {
    const float* Wq_l = Wq + (long)layer * HID_ * (NH_ * HD_);
    const float* Wk_l = Wk + (long)layer * HID_ * (NKV_ * HD_);
    const float* Wv_l = Wv + (long)layer * HID_ * (NKV_ * HD_);
    const float* Wo_l = Wo + (long)layer * (NH_ * HD_) * HID_;
    const float* Wg_l = Wg + (long)layer * HID_ * INTER_;
    const float* Wu_l = Wu + (long)layer * HID_ * INTER_;
    const float* Wd_l = Wd + (long)layer * INTER_ * HID_;

    // ---- attention block ----
    rms_kernel<<<MROWS_, 256, 0, stream>>>(x, ln1 + layer * HID_, hb16);

    transpose_bf16_kernel<<<dim3((NH_*HD_)/32, HID_/32), gB, 0, stream>>>(Wq_l, wtb, HID_, NH_*HD_);
    gemm_bf16_wmma<<<dim3((NH_*HD_)/64, gy), gB, 0, stream>>>(hb16, wtb, nullptr, qbf, MROWS_, NH_*HD_, HID_);
    transpose_bf16_kernel<<<dim3((NKV_*HD_)/32, HID_/32), gB, 0, stream>>>(Wk_l, wtb, HID_, NKV_*HD_);
    gemm_bf16_wmma<<<dim3((NKV_*HD_)/64, gy), gB, 0, stream>>>(hb16, wtb, nullptr, kbf, MROWS_, NKV_*HD_, HID_);
    transpose_bf16_kernel<<<dim3((NKV_*HD_)/32, HID_/32), gB, 0, stream>>>(Wv_l, wtb, HID_, NKV_*HD_);
    gemm_bf16_wmma<<<dim3((NKV_*HD_)/64, gy), gB, 0, stream>>>(hb16, wtb, nullptr, vbf, MROWS_, NKV_*HD_, HID_);

    rope_store_kernel<<<MROWS_, 64, 0, stream>>>(qbf, kbf, vbf, pos, qh, kh, vh,
                                                 keys_out + (long)layer * BPN,
                                                 vals_out + (long)layer * BPN);
    attn_kernel<<<dim3(16, NH_, B_), 128, ATTN_SMEM, stream>>>(qh, kh, vh, pad, ab16);

    transpose_bf16_kernel<<<dim3(HID_/32, (NH_*HD_)/32), gB, 0, stream>>>(Wo_l, wtb, NH_*HD_, HID_);
    gemm_bf16_wmma<<<dim3(HID_/64, gy), gB, 0, stream>>>(ab16, wtb, x, x, MROWS_, HID_, NH_*HD_);

    // ---- mlp block ----
    rms_kernel<<<MROWS_, 256, 0, stream>>>(x, ln2 + layer * HID_, hb16);

    transpose_bf16_kernel<<<dim3(INTER_/32, HID_/32), gB, 0, stream>>>(Wg_l, wtb, HID_, INTER_);
    gemm_bf16_wmma<<<dim3(INTER_/64, gy), gB, 0, stream>>>(hb16, wtb, nullptr, gbf, MROWS_, INTER_, HID_);
    transpose_bf16_kernel<<<dim3(INTER_/32, HID_/32), gB, 0, stream>>>(Wu_l, wtb, HID_, INTER_);
    gemm_bf16_wmma<<<dim3(INTER_/64, gy), gB, 0, stream>>>(hb16, wtb, nullptr, ubf, MROWS_, INTER_, HID_);

    silu_mul_kernel<<<(int)((SZ_I + 255) / 256), 256, 0, stream>>>(gbf, ubf, tb16, SZ_I);

    transpose_bf16_kernel<<<dim3(HID_/32, INTER_/32), gB, 0, stream>>>(Wd_l, wtb, INTER_, HID_);
    gemm_bf16_wmma<<<dim3(HID_/64, gy), gB, 0, stream>>>(tb16, wtb, x, x, MROWS_, HID_, INTER_);
  }
}
